// LSTM_62843961475296
// MI455X (gfx1250) — matrix operand order repaired and checked by reading, not compiled, not run on hardware
//
#include <hip/hip_runtime.h>
#include <math.h>

// ---- problem sizes (fixed by reference) ----
#define VOCAB 10
#define EMB   10
#define HDIM  1024
#define SEQ   512
#define BSZ   128
#define NCLS  10
#define N4H   4096          // 4*HDIM
#define KB    32            // K blocks of 32 (HDIM/32)
#define NWG   128           // 64 column-chunks x 2 M-halves
#define NTHR  256           // 8 wave32 per WG

#define LDSB_BYTES (4 * KB * 32 * 16 * 2)        // 131072: packed B slice (bf16)
#define LDSX_BYTES (4 * 2 * 32 * 8 * 4)          // 8192: f/o accumulator exchange (f32)

typedef __attribute__((ext_vector_type(16))) __bf16 v16bf;
typedef __attribute__((ext_vector_type(8)))  __bf16 v8bf;
typedef __attribute__((ext_vector_type(8)))  float  v8f;

__device__ __forceinline__ float fast_sigm(float x) {
    return __builtin_amdgcn_rcpf(1.0f + __expf(-x));
}
__device__ __forceinline__ float fast_tanh(float x) {
    return 2.0f * __builtin_amdgcn_rcpf(1.0f + __expf(-2.0f * x)) - 1.0f;
}

// ---------------------------------------------------------------------------
// Init: zero ping-pong h buffers + barrier counter, and build the token
// projection table  T[v][g*H + c] = bias_g[c] + sum_e emb[v][e] * Wx_g[e][c].
// (VOCAB=10 makes the whole input projection a 160 KB lookup table.)
// ---------------------------------------------------------------------------
__global__ void k_init(float* __restrict__ Ttab, __bf16* __restrict__ hbuf,
                       unsigned* __restrict__ bar,
                       const float* __restrict__ emb,
                       const float* __restrict__ Wgx, const float* __restrict__ Wix,
                       const float* __restrict__ Wfx, const float* __restrict__ Wox,
                       const float* __restrict__ bg,  const float* __restrict__ bi,
                       const float* __restrict__ bf_, const float* __restrict__ bo)
{
    const int t = blockIdx.x * blockDim.x + threadIdx.x;
    const int nt = gridDim.x * blockDim.x;
    if (t == 0) *bar = 0u;
    for (int i = t; i < 2 * BSZ * HDIM; i += nt) hbuf[i] = (__bf16)0.0f;
    for (int i = t; i < VOCAB * N4H; i += nt) {
        const int v = i / N4H, n = i % N4H;
        const int g = n >> 10, hc = n & (HDIM - 1);
        const float* Wx = (g == 0) ? Wgx : (g == 1) ? Wix : (g == 2) ? Wfx : Wox;
        const float* bb = (g == 0) ? bg  : (g == 1) ? bi  : (g == 2) ? bf_ : bo;
        float acc = bb[hc];
        #pragma unroll
        for (int e = 0; e < EMB; ++e) acc += emb[v * EMB + e] * Wx[e * HDIM + hc];
        Ttab[i] = acc;
    }
}

// ---------------------------------------------------------------------------
// Persistent LSTM kernel: 128 WGs x 8 waves.
//   blockIdx = (nc, mh): nc = 16-column chunk of H, mh = which 64-row half of B.
//   wave w:  mtile = w>>1 (rows mh*64 + mtile*16), gate pair gp = w&1
//            (gp==0 -> gates g,i ; gp==1 -> gates f,o).
// Weight slice (all 4 gates x 1024 K x 16 cols, bf16) is staged once into
// 128 KB LDS in the per-lane WMMA B layout. Odd waves pass their f/o
// accumulators to the paired even wave through an 8 KB LDS exchange; even
// waves hold the cell state in VGPRs for all 512 steps and do the fused
// gate math + bf16 h writeback. Timesteps are separated by a device-scope
// split grid barrier; the next step's token-table gathers (h-independent)
// are issued between barrier signal and barrier wait to hide its latency.
// ---------------------------------------------------------------------------
__global__ void __launch_bounds__(NTHR, 1) k_lstm(
    const int*   __restrict__ x,      // [B][SEQ]
    const float* __restrict__ Ttab,   // [VOCAB][4H]
    const float* __restrict__ Wgh, const float* __restrict__ Wih,
    const float* __restrict__ Wfh, const float* __restrict__ Woh,
    __bf16* __restrict__ hbuf,        // [2][B][H] ping-pong
    unsigned* __restrict__ bar)
{
    extern __shared__ char lds_raw[];
    __bf16* ldsB = (__bf16*)lds_raw;                    // [4][KB][32][16] bf16
    float*  ldsX = (float*)(lds_raw + LDSB_BYTES);      // [4 mtiles][2][32][8] f32

    const int nc    = blockIdx.x >> 1;    // 0..63
    const int mh    = blockIdx.x & 1;     // 0..1
    const int tid   = threadIdx.x;
    const int lane  = tid & 31;
    const int wave  = tid >> 5;           // 0..7
    const int mtile = wave >> 1;          // 0..3
    const int gp    = wave & 1;           // 0: g,i   1: f,o
    const int hfl   = lane >> 4;
    const int nl    = lane & 15;

    // ---- one-time stage of the weight slice into LDS (WMMA B layout):
    // entry (g, kb, lane, e) = Wh_g[kb*32 + (lane>=16)*16 + e][nc*16 + (lane&15)]
    for (int i = tid; i < 4 * KB * 32; i += NTHR) {
        const int g  = i >> 10;
        const int kb = (i >> 5) & 31;
        const int ln = i & 31;
        const int col   = nc * 16 + (ln & 15);
        const int kbase = kb * 32 + (ln >> 4) * 16;
        const float* W = (g == 0) ? Wgh : (g == 1) ? Wih : (g == 2) ? Wfh : Woh;
        __bf16* dst = &ldsB[(size_t)i * 16];
        #pragma unroll
        for (int e = 0; e < 16; ++e)
            dst[e] = (__bf16)W[(size_t)(kbase + e) * HDIM + col];
    }
    __syncthreads();

    const int mbase = mh * 64 + mtile * 16;
    const int mrow  = mbase + nl;                      // A-matrix row for this lane
    const int gbase = gp * 2;
    const __bf16* __restrict__ ldsBg0 = ldsB + (size_t)(gbase + 0) * KB * 32 * 16;
    const __bf16* __restrict__ ldsBg1 = ldsB + (size_t)(gbase + 1) * KB * 32 * 16;
    float* __restrict__ exch = ldsX + (size_t)mtile * 2 * 32 * 8;

    v8f c = {};                 // cell state (even waves), C layout: r -> row hfl*8+r
    unsigned barTarget = 0;

    // Token-table prefetch registers (even waves only): input proj + bias for
    // all 4 gates of this wave's 8 C-rows, next timestep.
    float tg[8], ti[8], tf[8], to[8];
    auto gatherT = [&](int s) {
        #pragma unroll
        for (int r = 0; r < 8; ++r) {
            const int m   = mbase + hfl * 8 + r;
            const int tok = x[m * SEQ + s];
            const float* Trow = Ttab + (size_t)tok * N4H + nc * 16 + nl;
            tg[r] = Trow[0 * HDIM];
            ti[r] = Trow[1 * HDIM];
            tf[r] = Trow[2 * HDIM];
            to[r] = Trow[3 * HDIM];
        }
    };
    if (gp == 0) gatherT(0);

    for (int s = 0; s < SEQ; ++s) {
        const __bf16* __restrict__ hin  = hbuf + (size_t)(s & 1) * BSZ * HDIM;
        __bf16*       __restrict__ hout = hbuf + (size_t)((s + 1) & 1) * BSZ * HDIM;

        v8f acc0 = {}, acc1 = {};
        const __bf16* __restrict__ arow = hin + (size_t)mrow * HDIM;

        for (int kb = 0; kb < KB; ++kb) {
            // A tile (16x32 bf16): lane holds K = kb*32 + hfl*16 + {0..15}
            v8bf alo = *(const v8bf*)(arow + kb * 32 + hfl * 16);
            v8bf ahi = *(const v8bf*)(arow + kb * 32 + hfl * 16 + 8);
            v16bf a;
            #pragma unroll
            for (int e = 0; e < 8; ++e) { a[e] = alo[e]; a[8 + e] = ahi[e]; }

            const __bf16* bb = (const __bf16*)(((size_t)kb * 32 + lane) * 16);
            v16bf b0 = *(const v16bf*)(ldsBg0 + ((size_t)kb * 32 + lane) * 16);
            v16bf b1 = *(const v16bf*)(ldsBg1 + ((size_t)kb * 32 + lane) * 16);
            (void)bb;

            acc0 = __builtin_amdgcn_wmma_f32_16x16x32_bf16(false, a, false, b0, (short)0, acc0, false, false);
            acc1 = __builtin_amdgcn_wmma_f32_16x16x32_bf16(false, a, false, b1, (short)0, acc1, false, false);
        }

        // ---- odd waves publish f/o accumulators for their paired even wave
        if (gp == 1) {
            *(v8f*)(exch + (size_t)(0 * 32 + lane) * 8) = acc0;   // gate f
            *(v8f*)(exch + (size_t)(1 * 32 + lane) * 8) = acc1;   // gate o
        }
        __syncthreads();

        // ---- even waves: fused epilogue (pregathered input proj + bias,
        //      gate nonlinearities, cell update, bf16 h writeback)
        if (gp == 0) {
            const v8f fa = *(const v8f*)(exch + (size_t)(0 * 32 + lane) * 8);
            const v8f oa = *(const v8f*)(exch + (size_t)(1 * 32 + lane) * 8);
            #pragma unroll
            for (int r = 0; r < 8; ++r) {
                const float ag = acc0[r] + tg[r];
                const float ai = acc1[r] + ti[r];
                const float af = fa[r]   + tf[r];
                const float ao = oa[r]   + to[r];
                const float cn = fast_tanh(ag) * fast_sigm(ai) + c[r] * fast_sigm(af);
                c[r] = cn;
                const int m = mbase + hfl * 8 + r;
                hout[(size_t)m * HDIM + nc * 16 + nl] = (__bf16)(fast_tanh(cn) * fast_sigm(ao));
            }
        }

        // ---- device-scope split grid barrier between timesteps.
        __threadfence();
        __syncthreads();
        barTarget += NWG;
        if (tid == 0)
            __hip_atomic_fetch_add(bar, 1u, __ATOMIC_RELEASE, __HIP_MEMORY_SCOPE_AGENT);
        // Overlap: next step's token-table gathers are independent of h.
        if (gp == 0 && s + 1 < SEQ) gatherT(s + 1);
        if (tid == 0)
            while (__hip_atomic_load(bar, __ATOMIC_ACQUIRE, __HIP_MEMORY_SCOPE_AGENT) < barTarget)
                __builtin_amdgcn_s_sleep(1);
        __syncthreads();
    }
}

// ---------------------------------------------------------------------------
// Final classifier: out[b][j] = h_T[b] . Wph[:,j] + bp[j]   (tiny: 128x10)
// ---------------------------------------------------------------------------
__global__ void k_proj(const __bf16* __restrict__ hT, const float* __restrict__ Wph,
                       const float* __restrict__ bp, float* __restrict__ out)
{
    const int t = blockIdx.x * blockDim.x + threadIdx.x;
    if (t >= BSZ * NCLS) return;
    const int b = t / NCLS, j = t % NCLS;
    float acc = bp[j];
    const __bf16* hr = hT + (size_t)b * HDIM;
    for (int k = 0; k < HDIM; ++k) acc += (float)hr[k] * Wph[k * NCLS + j];
    out[t] = acc;
}

// ---------------------------------------------------------------------------
extern "C" void kernel_launch(void* const* d_in, const int* in_sizes, int n_in,
                              void* d_out, int out_size, void* d_ws, size_t ws_size,
                              hipStream_t stream)
{
    const int*   x   = (const int*)  d_in[0];
    const float* emb = (const float*)d_in[1];
    const float* Wgx = (const float*)d_in[2];
    const float* Wgh = (const float*)d_in[3];
    const float* bg  = (const float*)d_in[4];
    const float* Wix = (const float*)d_in[5];
    const float* Wih = (const float*)d_in[6];
    const float* bi  = (const float*)d_in[7];
    const float* Wfx = (const float*)d_in[8];
    const float* Wfh = (const float*)d_in[9];
    const float* bf_ = (const float*)d_in[10];
    const float* Wox = (const float*)d_in[11];
    const float* Woh = (const float*)d_in[12];
    const float* bo  = (const float*)d_in[13];
    const float* Wph = (const float*)d_in[14];
    const float* bp  = (const float*)d_in[15];

    char* ws = (char*)d_ws;
    unsigned* bar  = (unsigned*)ws;                       // 256 B slot
    float*    Ttab = (float*)(ws + 256);                  // 10*4096*4 = 160 KB
    __bf16*   hbuf = (__bf16*)(ws + 256 + VOCAB * N4H * sizeof(float)); // 512 KB ping-pong

    k_init<<<160, 256, 0, stream>>>(Ttab, hbuf, bar, emb,
                                    Wgx, Wix, Wfx, Wox, bg, bi, bf_, bo);

    const size_t ldsBytes = (size_t)LDSB_BYTES + LDSX_BYTES;   // 136 KB
    k_lstm<<<NWG, NTHR, ldsBytes, stream>>>(x, Ttab, Wgh, Wih, Wfh, Woh, hbuf, bar);

    // SEQ is even -> final h lands in buffer 0
    k_proj<<<(BSZ * NCLS + 255) / 256, 256, 0, stream>>>(hbuf, Wph, bp, (float*)d_out);
}
